// GraphMambaLayer_62397284876372
// MI455X (gfx1250) — matrix operand order repaired and checked by reading, not compiled, not run on hardware
//
#include <hip/hip_runtime.h>
#include <hip/hip_bf16.h>

// GraphMambaLayer forward for MI455X (gfx1250, wave32, WMMA + TDM).
// Pipeline: prep weights (f16 transposed) -> zero -> edge scatter atomics ->
// stage1 (graph_proj GEMM + LayerNorm) -> stage2 (TDM tile load, in_proj GEMM,
// causal conv + fused SSM, out_proj GEMM, gate GEMM, sigmoid blend).
// GEMMs via v_wmma_f32_16x16x32_f16 (f32 accumulation); hln tile staged into
// LDS with tensor_load_to_lds (TENSORcnt).

typedef _Float16 h16;
typedef __attribute__((ext_vector_type(16))) _Float16 v16h;
typedef __attribute__((ext_vector_type(8)))  float    v8f;
typedef __attribute__((ext_vector_type(4)))  unsigned int u32x4;
typedef __attribute__((ext_vector_type(8)))  int i32x8;
typedef __attribute__((ext_vector_type(4)))  int i32x4;

#define NTHREADS 256   // 8 waves of 32

// ---------------- WMMA fragment loaders (contiguous k order: k = 16*(lane/16) + s) ----------------

// A fragment: 16x32 f16 tile. sa is row-major f16 with leading dim lda (halves).
// lane<16: halves[0..15]=K(kb..kb+15)
// lane>=16: +16 on K. Row m = mbase + (lane&15).
__device__ __forceinline__ v16h load_frag_a(const h16* sa, int lda, int mbase,
                                            int kb, int lane) {
  const h16* p = sa + (mbase + (lane & 15)) * lda + kb + ((lane & 16) ? 16 : 0);
  v16h r;
#pragma unroll
  for (int j = 0; j < 16; ++j) { r[j] = p[j]; }
  return r;
}

// B fragment: 32x16 f16 tile from transposed weight wt stored [N][K] (halves),
// so K is contiguous. lane holds column n = nbase+(lane&15);
// lane<16: K kb..kb+15, lane>=16: K kb+16..kb+31  -> one aligned 32B v16h load.
__device__ __forceinline__ v16h load_frag_b(const h16* wt, int K, int nbase,
                                            int kb, int lane) {
  const h16* p = wt + (nbase + (lane & 15)) * K + kb + ((lane & 16) ? 16 : 0);
  return *(const v16h*)p;
}

// GEMM: C[Mtiles*16 x Ntiles*16] = A[.. x Ksteps*32] * Wt^T (+bias), f32 out to LDS.
// 8 waves per block; tiles round-robined over waves.
__device__ __forceinline__ void wmma_gemm_f32(const h16* sa, int lda,
                                              const h16* wt, int Ksteps,
                                              int Mtiles, int Ntiles,
                                              float* sc, int ldc,
                                              const float* bias, int tid) {
  const int wave = tid >> 5, lane = tid & 31;
  const int K = Ksteps * 32;
  for (int t = wave; t < Mtiles * Ntiles; t += 8) {
    const int mt = t / Ntiles, nt = t % Ntiles;
    v8f acc = {0.f, 0.f, 0.f, 0.f, 0.f, 0.f, 0.f, 0.f};
    for (int ks = 0; ks < Ksteps; ++ks) {
      v16h a = load_frag_a(sa, lda, mt * 16, ks * 32, lane);
      v16h b = load_frag_b(wt, K, nt * 16, ks * 32, lane);
      acc = __builtin_amdgcn_wmma_f32_16x16x32_f16(false, a, false, b,
                                                   (short)0, acc, false, false);
    }
    const int n = nt * 16 + (lane & 15);
    const float bv = bias ? bias[n] : 0.f;
    const int mb = mt * 16 + ((lane & 16) ? 8 : 0);
#pragma unroll
    for (int v = 0; v < 8; ++v) sc[(mb + v) * ldc + n] = acc[v] + bv;
  }
}

// Same but converts the result to f16 (for staging the next GEMM's A operand).
__device__ __forceinline__ void wmma_gemm_f16(const h16* sa, int lda,
                                              const h16* wt, int Ksteps,
                                              int Mtiles, int Ntiles,
                                              h16* sc, int ldc,
                                              const float* bias, int tid) {
  const int wave = tid >> 5, lane = tid & 31;
  const int K = Ksteps * 32;
  for (int t = wave; t < Mtiles * Ntiles; t += 8) {
    const int mt = t / Ntiles, nt = t % Ntiles;
    v8f acc = {0.f, 0.f, 0.f, 0.f, 0.f, 0.f, 0.f, 0.f};
    for (int ks = 0; ks < Ksteps; ++ks) {
      v16h a = load_frag_a(sa, lda, mt * 16, ks * 32, lane);
      v16h b = load_frag_b(wt, K, nt * 16, ks * 32, lane);
      acc = __builtin_amdgcn_wmma_f32_16x16x32_f16(false, a, false, b,
                                                   (short)0, acc, false, false);
    }
    const int n = nt * 16 + (lane & 15);
    const float bv = bias ? bias[n] : 0.f;
    const int mb = mt * 16 + ((lane & 16) ? 8 : 0);
#pragma unroll
    for (int v = 0; v < 8; ++v) sc[(mb + v) * ldc + n] = (h16)(acc[v] + bv);
  }
}

// ---------------- Kernel 1: weight prep (f16 transposed) + bc --------------

__global__ void k_prep(const float* Wg, const float* Wgp, const float* Win,
                       const float* Wout, const float* Bm, const float* Cm,
                       h16* WtG, h16* WtGP, h16* WtIn, h16* WtOut, float* bc) {
  const int tid = blockIdx.x * blockDim.x + threadIdx.x;
  const int T = gridDim.x * blockDim.x;
  for (int i = tid; i < 128 * 128; i += T) {            // (K=128, N=128)
    int k = i >> 7, n = i & 127;
    WtG[n * 128 + k]  = (h16)Wg[i];
    WtGP[n * 128 + k] = (h16)Wgp[i];
  }
  for (int i = tid; i < 128 * 256; i += T) {            // W_in (K=128, N=256)
    int k = i >> 8, n = i & 255;
    WtIn[n * 128 + k] = (h16)Win[i];
  }
  for (int i = tid; i < 256 * 128; i += T) {            // W_out (K=256, N=128)
    int k = i >> 7, n = i & 127;
    WtOut[n * 256 + k] = (h16)Wout[i];
  }
  for (int d = tid; d < 256; d += T) {
    float s = 0.f;
#pragma unroll
    for (int j = 0; j < 16; ++j) s += Bm[d * 16 + j] * Cm[d * 16 + j];
    bc[d] = s;
  }
}

// ---------------- Kernel 2: zero agg + deg (contiguous) --------------------

__global__ void k_zero(float* p, int n) {
  const int tid = blockIdx.x * blockDim.x + threadIdx.x;
  const int T = gridDim.x * blockDim.x;
  for (int i = tid; i < n; i += T) p[i] = 0.f;
}

// ---------------- Kernel 3: edge scatter-add (degree-normalized later) -----

__global__ void k_scatter(const float* __restrict__ x,
                          const int* __restrict__ index,
                          float* agg, float* deg, int E) {
  const int e = blockIdx.x * 2 + (threadIdx.x >> 7);
  const int c = threadIdx.x & 127;
  if (e >= E) return;
  const int row = index[e];
  const int col = index[E + e];
  // pull the gathered row toward L2 (global_prefetch_b8)
  __builtin_prefetch(&x[row * 128 + c], 0, 1);
  unsafeAtomicAdd(&agg[col * 128 + c], x[row * 128 + c]);
  if (c == 0) unsafeAtomicAdd(&deg[col], 1.0f);
}

// ---------------- Kernel 4: graph_proj GEMM + residual + LayerNorm ---------
// 64 rows per block. hln (f16) written to ws for stage2.

__global__ void k_stage1(const float* __restrict__ x,
                         const float* __restrict__ agg,
                         const float* __restrict__ deg,
                         const h16* __restrict__ WtGP,
                         const float* __restrict__ b_gp,
                         const float* __restrict__ gamma,
                         const float* __restrict__ beta,
                         h16* __restrict__ hln) {
  __shared__ __align__(32) h16 sA[64 * 128];   // agg/deg, f16 (16 KB)
  __shared__ __align__(16) float sH[64 * 128]; // h, f32 (32 KB)
  __shared__ float redS[256], redQ[256];

  const int t0 = blockIdx.x * 64;
  const int tid = threadIdx.x;

  for (int i = tid; i < 64 * 128; i += NTHREADS) {
    const int r = i >> 7, c = i & 127;
    const int node = t0 + r;
    float dv = deg[node];
    dv = dv > 1.f ? dv : 1.f;
    sA[i] = (h16)(agg[node * 128 + c] / dv);
  }
  __syncthreads();

  // h_tile = aggn @ W_gp + b_gp   (64x128 x 128x128)
  wmma_gemm_f32(sA, 128, WtGP, /*Ksteps=*/4, /*Mtiles=*/4, /*Ntiles=*/8,
                sH, 128, b_gp, tid);
  __syncthreads();

  // add x, per-row mean/var (4 threads per row, 32 cols each)
  const int r = tid >> 2, q = tid & 3;
  const int node = t0 + r;
  float s1 = 0.f, s2 = 0.f;
  for (int c = q * 32; c < q * 32 + 32; ++c) {
    float v = sH[r * 128 + c] + x[node * 128 + c];
    sH[r * 128 + c] = v;
    s1 += v; s2 += v * v;
  }
  redS[tid] = s1; redQ[tid] = s2;
  __syncthreads();
  if (q == 0) {
    float a = redS[tid] + redS[tid + 1] + redS[tid + 2] + redS[tid + 3];
    float b = redQ[tid] + redQ[tid + 1] + redQ[tid + 2] + redQ[tid + 3];
    float mu = a * (1.f / 128.f);
    float var = b * (1.f / 128.f) - mu * mu;
    redS[tid] = mu;
    redQ[tid] = rsqrtf(var + 1e-5f);
  }
  __syncthreads();
  const float mu = redS[r * 4], rstd = redQ[r * 4];
  for (int c = q * 32; c < q * 32 + 32; ++c) {
    float v = (sH[r * 128 + c] - mu) * rstd * gamma[c] + beta[c];
    hln[node * 128 + c] = (h16)v;
  }
}

// ---------------- Kernel 5: in_proj + conv + SSM + out_proj + gate + blend -
// 32 output rows per block, 16-row halo for the causal depthwise conv (k=4).
// hln tile staged into LDS via the Tensor Data Mover (TENSORcnt).
// Phased LDS region reuse, 52 KB static total.

__global__ void k_stage2(const float* __restrict__ x,
                         const h16* __restrict__ hln,
                         const h16* __restrict__ WtIn,
                         const float* __restrict__ b_in,
                         const float* __restrict__ conv_w,
                         const float* __restrict__ conv_b,
                         const float* __restrict__ bc,
                         const float* __restrict__ Dp,
                         const h16* __restrict__ WtOut,
                         const float* __restrict__ b_out,
                         const h16* __restrict__ WtG,
                         const float* __restrict__ b_gate,
                         float* __restrict__ out) {
  __shared__ __align__(32) char sm[53248];     // 52 KB, phased reuse; only
                                               // shared object -> LDS offset 0
  h16*   sHln  = (h16*)(sm);                   //  0    .. 12288 : 48x128 f16
  h16*   sU    = (h16*)(sm + 12288);           // 12288 .. 36864 : 48x256 f16
  h16*   sY    = (h16*)(sm + 36864);           // 36864 .. 53248 : 32x256 f16
  float* sOut  = (float*)(sm);                 //  0    .. 16384 : 32x128 f32
  h16*   sX    = (h16*)(sm + 16384);           // 16384 .. 24576 : 32x128 f16
  float* sGate = (float*)(sm + 24576);         // 24576 .. 40960 : 32x128 f32

  const int t0 = blockIdx.x * 32;
  const int tid = threadIdx.x;

  // P1: TDM-stage hln rows [t0-16, t0+32) into sHln.
  // 2D descriptor: data_size=2B, tile 128 x 48, tensor_dim0_stride=128 elems.
  if ((tid >> 5) == 0) {                       // one wave issues the TDM op
    const unsigned long long ga =
        (unsigned long long)(uintptr_t)(hln + (long long)(t0 - 16) * 128);
    u32x4 g0;
    g0[0] = 1u;                                        // count=1, user mode
    g0[1] = 0u;                                        // lds_addr = sHln = 0
    g0[2] = (unsigned)(ga & 0xFFFFFFFFu);              // global_addr[31:0]
    g0[3] = (unsigned)((ga >> 32) & 0x01FFFFFFu)       // global_addr[56:32]
            | (2u << 30);                              // type=2 ("image")
    i32x8 g1;
    g1[0] = (int)(1u << 16);                           // data_size=1 (2 bytes)
    g1[1] = (int)(128u << 16);                         // tensor_dim0[15:0]
    g1[2] = (int)(48u << 16);                          // td0 hi=0 | tensor_dim1 lo
    g1[3] = (int)(128u << 16);                         // td1 hi=0 | tile_dim0=128
    g1[4] = 48;                                        // tile_dim1=48, tile_dim2=0
    g1[5] = 128;                                       // tensor_dim0_stride lo
    g1[6] = 0;                                         // stride hi / dim1_stride
    g1[7] = 0;
    i32x4 gz4 = {0, 0, 0, 0};                          // 2D: groups 2/3 unused
    i32x8 gz8 = {0, 0, 0, 0, 0, 0, 0, 0};              // 6-arg form: extra group
    __builtin_amdgcn_tensor_load_to_lds(g0, g1, gz4, gz4, gz8, 0);
    __builtin_amdgcn_s_wait_tensorcnt(0);
  }
  __syncthreads();
  if (t0 == 0) {   // halo rows before node 0 must be zero (conv zero-padding)
    for (int i = tid; i < 16 * 128; i += NTHREADS) sHln[i] = (h16)0.f;
    __syncthreads();
  }

  // P2: u = hln @ W_in + b_in   (48x128 x 128x256) -> f16 LDS
  wmma_gemm_f16(sHln, 128, WtIn, /*Ksteps=*/4, /*Mtiles=*/3, /*Ntiles=*/16,
                sU, 256, b_in, tid);
  __syncthreads();

  // P3: causal depthwise conv (window 4) + fused SSM: y = v^2*bc + Dp*v
  for (int i = tid; i < 32 * 256; i += NTHREADS) {
    const int r = i >> 8, d = i & 255;
    const int m = t0 + r;
    float v = conv_b[d];
#pragma unroll
    for (int k = 0; k < 4; ++k) {
      const int node = m - 3 + k;                    // zero-padded before node 0
      const float uv = (node >= 0) ? (float)sU[(r + 13 + k) * 256 + d] : 0.f;
      v += conv_w[d * 4 + k] * uv;
    }
    const float y = v * v * bc[d] + Dp[d] * v;
    sY[i] = (h16)y;
  }
  __syncthreads();

  // P4: out = y @ W_out + b_out (32x256 x 256x128); also stage x tile as f16.
  wmma_gemm_f32(sY, 256, WtOut, /*Ksteps=*/8, /*Mtiles=*/2, /*Ntiles=*/8,
                sOut, 128, b_out, tid);
  for (int i = tid; i < 32 * 128; i += NTHREADS)
    sX[i] = (h16)x[t0 * 128 + i];
  __syncthreads();

  // P5: gate = x @ W_gate + b_gate
  wmma_gemm_f32(sX, 128, WtG, /*Ksteps=*/4, /*Mtiles=*/2, /*Ntiles=*/8,
                sGate, 128, b_gate, tid);
  __syncthreads();

  // P6: final = sigmoid(gate)*out + (1-sigmoid(gate))*residual
  for (int i = tid; i < 32 * 128; i += NTHREADS) {
    const float g = 1.f / (1.f + expf(-sGate[i]));
    const float res = x[t0 * 128 + i];
    out[t0 * 128 + i] = g * sOut[i] + (1.f - g) * res;
  }
}

// ---------------- host-side launcher ---------------------------------------

extern "C" void kernel_launch(void* const* d_in, const int* in_sizes, int n_in,
                              void* d_out, int out_size, void* d_ws, size_t ws_size,
                              hipStream_t stream) {
  const float* x      = (const float*)d_in[0];
  const int*   index  = (const int*)d_in[1];
  // d_in[2] = attr (unused by forward)
  const float* W_gate = (const float*)d_in[3];
  const float* b_gate = (const float*)d_in[4];
  const float* W_gp   = (const float*)d_in[5];
  const float* b_gp   = (const float*)d_in[6];
  const float* ln_g   = (const float*)d_in[7];
  const float* ln_b   = (const float*)d_in[8];
  const float* W_in   = (const float*)d_in[9];
  const float* b_in   = (const float*)d_in[10];
  const float* conv_w = (const float*)d_in[11];
  const float* conv_b = (const float*)d_in[12];
  // d_in[13] = A (unused: zero-init state kills the exp(x*A) term)
  const float* Bm     = (const float*)d_in[14];
  const float* Cm     = (const float*)d_in[15];
  const float* Dp     = (const float*)d_in[16];
  const float* W_out  = (const float*)d_in[17];
  const float* b_out  = (const float*)d_in[18];
  float* out = (float*)d_out;

  const int Nn = in_sizes[0] / 128;       // 65536 nodes
  const int E  = in_sizes[1] / 2;         // 524288 edges

  // workspace layout (bytes)
  char* ws = (char*)d_ws;
  float* agg   = (float*)(ws);                       // Nn*128 f32 = 32 MB
  float* deg   = (float*)(ws + (size_t)Nn * 128 * 4);// Nn f32 (contiguous w/ agg)
  size_t off   = (size_t)Nn * 129 * 4;
  off = (off + 1023) & ~(size_t)1023;
  h16* hln     = (h16*)(ws + off);  off += (size_t)Nn * 128 * 2;   // 16 MB
  off = (off + 1023) & ~(size_t)1023;
  h16* WtG     = (h16*)(ws + off);  off += 128 * 128 * 2;
  h16* WtGP    = (h16*)(ws + off);  off += 128 * 128 * 2;
  h16* WtIn    = (h16*)(ws + off);  off += 256 * 128 * 2;
  h16* WtOut   = (h16*)(ws + off);  off += 128 * 256 * 2;
  float* bc    = (float*)(ws + off);

  // 1) weight prep (transposed f16) + bc
  k_prep<<<64, NTHREADS, 0, stream>>>(W_gate, W_gp, W_in, W_out, Bm, Cm,
                                      WtG, WtGP, WtIn, WtOut, bc);
  // 2) zero agg + deg (contiguous Nn*129 floats)
  k_zero<<<2048, NTHREADS, 0, stream>>>(agg, Nn * 129);
  // 3) edge scatter atomics
  k_scatter<<<(E + 1) / 2, NTHREADS, 0, stream>>>(x, index, agg, deg, E);
  // 4) graph_proj + LN -> hln
  k_stage1<<<Nn / 64, NTHREADS, 0, stream>>>(x, agg, deg, WtGP, b_gp,
                                             ln_g, ln_b, hln);
  // 5) in_proj + conv + SSM + out_proj + gate + blend
  k_stage2<<<Nn / 32, NTHREADS, 0, stream>>>(x, hln, WtIn, b_in, conv_w, conv_b,
                                             bc, Dp, WtOut, b_out, WtG, b_gate,
                                             out);
}
